// GraphCAGE_6914897347017
// MI455X (gfx1250) — matrix operand order repaired and checked by reading, not compiled, not run on hardware
//
#include <hip/hip_runtime.h>

// ---------------------------------------------------------------------------
// GraphCAGE forward on gfx1250 (MI455X). B=64, N=512, D=128.
// Strategy: one up-front f32->bf16 conversion/transpose pass, then all GEMMs
// run v_wmma_f32_16x16x32_bf16 with pure b128 operand loads (no pack VALU in
// the inner loop). Routing is an LDS-reduction kernel.
// ---------------------------------------------------------------------------

typedef __attribute__((ext_vector_type(16))) __bf16       v16bf;
typedef __attribute__((ext_vector_type(8)))  float        v8f;
typedef __attribute__((ext_vector_type(8)))  unsigned int v8u;
typedef __attribute__((ext_vector_type(4)))  unsigned int v4u;

__device__ __forceinline__ unsigned int bf16_bits(float f) {
    unsigned int u = __float_as_uint(f);
    u += 0x7FFFu + ((u >> 16) & 1u);     // round-to-nearest-even
    return u >> 16;
}
__device__ __forceinline__ unsigned int pk_bf16(float lo, float hi) {
    return bf16_bits(lo) | (bf16_bits(hi) << 16);
}
__device__ __forceinline__ float bf2f(unsigned short u) {
    return __uint_as_float(((unsigned int)u) << 16);
}

// ---------------------------------------------------------------------------
// Elementwise f32 -> bf16 (n must be a multiple of 4).
// ---------------------------------------------------------------------------
__global__ void __launch_bounds__(256)
cvt_f32_to_bf16(const float* __restrict__ in, unsigned short* __restrict__ out,
                long long nQuads)
{
    long long i      = (long long)blockIdx.x * blockDim.x + threadIdx.x;
    long long stride = (long long)gridDim.x * blockDim.x;
    for (; i < nQuads; i += stride) {
        float4 f = *(const float4*)(in + i * 4);
        uint2  p;
        p.x = pk_bf16(f.x, f.y);
        p.y = pk_bf16(f.z, f.w);
        *(uint2*)(out + i * 4) = p;
    }
}

// ---------------------------------------------------------------------------
// Batched tiled transpose + convert: in [bat][R][C] f32 -> out [bat][C][R] bf16.
// R, C multiples of 32. Block = (32, 8).
// ---------------------------------------------------------------------------
__global__ void __launch_bounds__(256)
transcvt_f32_to_bf16(const float* __restrict__ in, unsigned short* __restrict__ out,
                     int R, int C)
{
    __shared__ float tile[32][33];
    const int bat = blockIdx.z;
    const int c0  = blockIdx.x * 32, r0 = blockIdx.y * 32;
    const float*    ip = in  + (long long)bat * R * C;
    unsigned short* op = out + (long long)bat * R * C;
    for (int j = threadIdx.y; j < 32; j += 8)
        tile[j][threadIdx.x] = ip[(long long)(r0 + j) * C + (c0 + threadIdx.x)];
    __syncthreads();
    for (int j = threadIdx.y; j < 32; j += 8)
        op[(long long)(c0 + j) * R + (r0 + threadIdx.x)] =
            (unsigned short)bf16_bits(tile[threadIdx.x][j]);
}

// ---------------------------------------------------------------------------
// Batched bf16 GEMM via WMMA:  C[b] = act( A[b] * B[b] + bias ),  Nc = 128.
//   A  element (m,k) = A [b*aBatch + m*aRow + k]        (row-major bf16)
//   B  element (k,n) = Bt[b*bBatch + n*bRow + k]        (B stored TRANSPOSED)
//   C  cTrans=0: Cb[m*cRow + n]   cTrans=1: Cb[n*cRow + m] (16B packed store)
// Block = 256 threads = 8 waves; wave w owns column tile w. grid=(M/16, bat).
// Inner loop per wave: 4x global_load_b128 + 1x v_wmma_f32_16x16x32_bf16.
// ---------------------------------------------------------------------------
__global__ void __launch_bounds__(256)
gemm_bf16_wmma(const unsigned short* __restrict__ A, long long aBatch, long long aRow,
               const unsigned short* __restrict__ Bt, long long bBatch, long long bRow,
               const float* __restrict__ bias,
               unsigned short* __restrict__ C, long long cBatch, long long cRow, int cTrans,
               int K, int act)
{
    const int lane  = threadIdx.x & 31;
    const int wave  = threadIdx.x >> 5;
    const int mBase = blockIdx.x * 16;
    const int batch = blockIdx.y;
    const int half  = lane >> 4;
    const int l15   = lane & 15;
    const int nCol  = wave * 16 + l15;

    const unsigned short* Arow =
        A  + (long long)batch * aBatch + (long long)(mBase + l15) * aRow;
    const unsigned short* Brow =
        Bt + (long long)batch * bBatch + (long long)nCol * bRow;

    v8f acc = {};
    for (int kk = 0; kk < K; kk += 32) {
        // A 16x32 frag: V0..3 = K half*8+0..7, V4..7 = K 16+half*8+0..7
        v4u a0 = *(const v4u*)(Arow + kk + half * 8);
        v4u a1 = *(const v4u*)(Arow + kk + half * 8 + 16);
        // B 32x16 frag: lane column nCol; V0..3 = K half*8+0..7, V4..7 = K 16+half*8+0..7
        v4u b0 = *(const v4u*)(Brow + kk + half * 8);
        v4u b1 = *(const v4u*)(Brow + kk + half * 8 + 16);
        v8u ua, ub;
        ua[0] = a0[0]; ua[1] = a0[1]; ua[2] = a0[2]; ua[3] = a0[3];
        ua[4] = a1[0]; ua[5] = a1[1]; ua[6] = a1[2]; ua[7] = a1[3];
        ub[0] = b0[0]; ub[1] = b0[1]; ub[2] = b0[2]; ub[3] = b0[3];
        ub[4] = b1[0]; ub[5] = b1[1]; ub[6] = b1[2]; ub[7] = b1[3];
        acc = __builtin_amdgcn_wmma_f32_16x16x32_bf16(
                  false, __builtin_bit_cast(v16bf, ua),
                  false, __builtin_bit_cast(v16bf, ub),
                  (short)0, acc, false, false);
    }

    const float bvn = bias ? bias[nCol] : 0.0f;
    unsigned short* Cb = C + (long long)batch * cBatch;
    if (cTrans) {
        float v[8];
#pragma unroll
        for (int r = 0; r < 8; ++r) {
            float t = acc[r] + bvn;
            v[r] = act ? tanhf(t) : t;
        }
        v4u st;
        st[0] = pk_bf16(v[0], v[1]); st[1] = pk_bf16(v[2], v[3]);
        st[2] = pk_bf16(v[4], v[5]); st[3] = pk_bf16(v[6], v[7]);
        *(v4u*)(Cb + (long long)nCol * cRow + mBase + half * 8) = st;
    } else {
#pragma unroll
        for (int r = 0; r < 8; ++r) {
            int   m = mBase + half * 8 + r;
            float t = acc[r] + bvn;
            if (act) t = tanhf(t);
            Cb[(long long)m * cRow + nCol] = (unsigned short)bf16_bits(t);
        }
    }
}

// ---------------------------------------------------------------------------
// Capsule dynamic routing. One 128-thread block per batch element.
// pc: [B, N, D] bf16. out[b, colOff:colOff+D] f32, row stride 6*D = 768.
// ---------------------------------------------------------------------------
__global__ void __launch_bounds__(128)
route_kernel(const unsigned short* __restrict__ pc, const int* __restrict__ pRouting,
             float* __restrict__ out, int colOff)
{
    const int N = 512, D = 128;
    const int b   = blockIdx.x;
    const int tid = threadIdx.x;
    const unsigned short* P = pc + (long long)b * N * D;

    __shared__ float bl[512];
    __shared__ float rc[512];
    __shared__ float lg[128];
    __shared__ float sred[128];

    for (int i = tid; i < N; i += 128) bl[i] = 0.0f;
    __syncthreads();

    const int R = *pRouting;
    for (int it = 0; it <= R; ++it) {
        // softmax(bl) -> rc over n
        float mloc = -3.0e38f;
        for (int i = tid; i < N; i += 128) mloc = fmaxf(mloc, bl[i]);
        sred[tid] = mloc; __syncthreads();
        for (int s = 64; s; s >>= 1) {
            if (tid < s) sred[tid] = fmaxf(sred[tid], sred[tid + s]);
            __syncthreads();
        }
        const float mx = sred[0]; __syncthreads();

        float sloc = 0.0f;
        for (int i = tid; i < N; i += 128) {
            float e = __expf(bl[i] - mx);
            rc[i] = e; sloc += e;
        }
        sred[tid] = sloc; __syncthreads();
        for (int s = 64; s; s >>= 1) {
            if (tid < s) sred[tid] += sred[tid + s];
            __syncthreads();
        }
        const float inv = 1.0f / sred[0]; __syncthreads();
        for (int i = tid; i < N; i += 128) rc[i] *= inv;
        __syncthreads();

        // logit[d] = tanh( sum_n pc[n,d]*rc[n] ),  d = tid
        float accd = 0.0f;
        for (int n = 0; n < N; ++n) accd += bf2f(P[(long long)n * D + tid]) * rc[n];
        const float t = tanhf(accd);

        if (it == R) {
            out[(long long)b * 768 + colOff + tid] = t;
        } else {
            lg[tid] = t; __syncthreads();
            for (int n = tid; n < N; n += 128) {   // bl[n] += sum_d pc[n,d]*logit[d]
                const unsigned short* row = P + (long long)n * D;
                float s2 = 0.0f;
                for (int d2 = 0; d2 < D; ++d2) s2 += bf2f(row[d2]) * lg[d2];
                bl[n] += s2;
            }
            __syncthreads();
        }
    }
}

// ---------------------------------------------------------------------------
extern "C" void kernel_launch(void* const* d_in, const int* in_sizes, int n_in,
                              void* d_out, int out_size, void* d_ws, size_t ws_size,
                              hipStream_t stream)
{
    const int B = 64, N = 512, D = 128;
    const long long ND  = (long long)N * D;
    const long long NN  = (long long)N * N;
    const long long DD  = (long long)D * D;
    const long long BND = (long long)B * ND;     // 4.19M
    const long long BNN = (long long)B * NN;     // 16.78M

    const float* x[3]   = {(const float*)d_in[0], (const float*)d_in[1], (const float*)d_in[2]};
    const float* adj[3] = {(const float*)d_in[3], (const float*)d_in[4], (const float*)d_in[5]};
    const float* c1w = (const float*)d_in[6];
    const float* c1b = (const float*)d_in[7];
    const float* c2w = (const float*)d_in[8];
    const float* c2b = (const float*)d_in[9];
    const float* wof = (const float*)d_in[10];
    const float* wos = (const float*)d_in[11];
    const float* wfp = (const float*)d_in[12];
    const float* wsp = (const float*)d_in[13];
    const int*   routing = (const int*)d_in[15];
    float* out = (float*)d_out;

    // ---- workspace layout (bf16 units) ----
    const size_t needU = (size_t)(BNN + 5 * BND + 2 * N * DD + 4 * DD);
    if (ws_size < needU * sizeof(unsigned short)) return;
    unsigned short* w    = (unsigned short*)d_ws;
    unsigned short* adjbf = w;  w += BNN;       // per-branch adjacency, reused
    unsigned short* Tx   = w;   w += BND;
    unsigned short* T0   = w;   w += BND;
    unsigned short* T1   = w;   w += BND;
    unsigned short* T2   = w;   w += BND;
    w += BND;                                   // slack / alignment headroom
    unsigned short* wfpT = w;   w += N * DD;    // W_fir_pc transposed per vertex
    unsigned short* wspT = w;   w += N * DD;
    unsigned short* w1bf = w;   w += DD;        // conv1_w (already B^T layout)
    unsigned short* w2bf = w;   w += DD;
    unsigned short* wofT = w;   w += DD;        // Wo_fir transposed
    unsigned short* wosT = w;   w += DD;

    const dim3 blk(256), tblk(32, 8);

    // ---- one-time weight conversion ----
    cvt_f32_to_bf16<<<dim3(16), blk, 0, stream>>>(c1w, w1bf, DD / 4);
    cvt_f32_to_bf16<<<dim3(16), blk, 0, stream>>>(c2w, w2bf, DD / 4);
    transcvt_f32_to_bf16<<<dim3(4, 4, 1),   tblk, 0, stream>>>(wof, wofT, D, D);
    transcvt_f32_to_bf16<<<dim3(4, 4, 1),   tblk, 0, stream>>>(wos, wosT, D, D);
    transcvt_f32_to_bf16<<<dim3(4, 4, 512), tblk, 0, stream>>>(wfp, wfpT, D, D);
    transcvt_f32_to_bf16<<<dim3(4, 4, 512), tblk, 0, stream>>>(wsp, wspT, D, D);

    for (int br = 0; br < 3; ++br) {
        const int c0 = (2 * br) * D, c1 = (2 * br + 1) * D;
        cvt_f32_to_bf16<<<dim3(1024), blk, 0, stream>>>(x[br],  Tx,    BND / 4);
        cvt_f32_to_bf16<<<dim3(2048), blk, 0, stream>>>(adj[br], adjbf, BNN / 4);

        // 1) hT[b][d][n] = (x @ W1^T + b1)^T            (batch = b, M = N)
        gemm_bf16_wmma<<<dim3(N / 16, B), blk, 0, stream>>>(
            Tx, ND, D,   w1bf, 0, D,   c1b,   T0, ND, N, 1,   D, 0);
        // 2) agg1[b][n][d] = adj @ h                    (K = N = 512)
        gemm_bf16_wmma<<<dim3(N / 16, B), blk, 0, stream>>>(
            adjbf, NN, N,   T0, ND, N,   nullptr,   T1, ND, D, 0,   N, 0);
        // 3) fir = tanh(agg1 @ Wo_fir)                  (M = B*N, 1 batch)
        gemm_bf16_wmma<<<dim3((B * N) / 16, 1), blk, 0, stream>>>(
            T1, 0, D,   wofT, 0, D,   nullptr,   T0, 0, D, 0,   D, 1);
        // 4) pc_fir[b,n,e] = fir[b,n,:] @ W_fir_pc[n]   (batch = vertex n)
        gemm_bf16_wmma<<<dim3(B / 16, N), blk, 0, stream>>>(
            T0, D, ND,   wfpT, DD, D,   nullptr,   T1, D, ND, 0,   D, 0);
        route_kernel<<<dim3(B), dim3(128), 0, stream>>>(T1, routing, out, c0);

        // 6) h2T = (fir @ W2^T + b2)^T
        gemm_bf16_wmma<<<dim3(N / 16, B), blk, 0, stream>>>(
            T0, ND, D,   w2bf, 0, D,   c2b,   T1, ND, N, 1,   D, 0);
        // 7) agg2 = adj @ h2
        gemm_bf16_wmma<<<dim3(N / 16, B), blk, 0, stream>>>(
            adjbf, NN, N,   T1, ND, N,   nullptr,   T2, ND, D, 0,   N, 0);
        // 8) sec = tanh(agg2 @ Wo_sec)
        gemm_bf16_wmma<<<dim3((B * N) / 16, 1), blk, 0, stream>>>(
            T2, 0, D,   wosT, 0, D,   nullptr,   T1, 0, D, 0,   D, 1);
        // 9) pc_sec
        gemm_bf16_wmma<<<dim3(B / 16, N), blk, 0, stream>>>(
            T1, D, ND,   wspT, DD, D,   nullptr,   T2, D, ND, 0,   D, 0);
        route_kernel<<<dim3(B), dim3(128), 0, stream>>>(T2, routing, out, c1);
    }
}
